// TransformerBlock_32323923869789
// MI455X (gfx1250) — matrix operand order repaired and checked by reading, not compiled, not run on hardware
//
#include <hip/hip_runtime.h>
#include <hip/hip_bf16.h>

typedef __attribute__((ext_vector_type(16))) _Float16 v16h;
typedef __attribute__((ext_vector_type(8)))  _Float16 v8h;
typedef __attribute__((ext_vector_type(8)))  float    v8f;

#define SEQ   2048
#define NB    8
#define DM    512
#define NH    8
#define DK    64
#define DFF   2048
#define MROWS (SEQ * NB)   // 16384

static __device__ inline v8f wmma32(v16h a, v16h b, v8f c) {
  return __builtin_amdgcn_wmma_f32_16x16x32_f16(
      false, a, false, b, (short)0, c, false, false);
}

// A-operand (16xK tile, M = lane&15). Layout: lanes 0-15 hold K {0..15},
// lanes 16-31 hold K {16..31}.
static __device__ inline v16h load_a_tile(const _Float16* base, int ld) {
  int l = threadIdx.x & 31;
  const _Float16* p = base + (l & 15) * ld + ((l >> 4) & 1) * 16;
  v8h lo = *(const v8h*)(p);
  v8h hi = *(const v8h*)(p + 8);
  v16h r;
#pragma unroll
  for (int i = 0; i < 8; ++i) { r[i] = lo[i]; r[i + 8] = hi[i]; }
  return r;
}

// B-operand (Kx16 tile) loaded from (N,K) K-contiguous storage:
// lane n = l&15, K-run of 16 at +16*(l>>4). One 32-byte load.
static __device__ inline v16h load_b_tile(const _Float16* base, int ld) {
  int l = threadIdx.x & 31;
  const _Float16* p = base + (l & 15) * ld + ((l >> 4) & 1) * 16;
  return *(const v16h*)(p);
}

// ---------------- packing / conversion ----------------

__global__ void pack_x_kernel(const float* __restrict__ x, _Float16* __restrict__ xh) {
  int idx = blockIdx.x * blockDim.x + threadIdx.x;     // [0, MROWS*DM)
  int d = idx & (DM - 1);
  int m = idx >> 9;
  int b = m >> 11, s = m & (SEQ - 1);
  xh[idx] = (_Float16)x[((size_t)s * NB + b) * DM + d];
}

__global__ void cvt_kernel(const float* __restrict__ in, _Float16* __restrict__ out, int n) {
  int idx = blockIdx.x * blockDim.x + threadIdx.x;
  if (idx < n) out[idx] = (_Float16)in[idx];
}

// in: (R,C) row-major f32 -> out: (C,R) row-major f16
__global__ void cvtT_kernel(const float* __restrict__ in, _Float16* __restrict__ out,
                            int R, int C) {
  int idx = blockIdx.x * blockDim.x + threadIdx.x;
  if (idx < R * C) {
    int r = idx / C, c = idx % C;
    out[(size_t)c * R + r] = (_Float16)in[idx];
  }
}

// ---------------- generic NT GEMM: C[m,n] = sum_k A[m,k]*W[n,k] + bias[n] ---------
// 256 threads = 8 waves; wave computes a 16(M) x 64(N) tile.
// B tile (64N x 32K) is staged once per block into a double-buffered LDS slab.
// Staging is two-stage pipelined: the global load for tile k+2 is issued at
// step k, and its data is ds_store'd at step k+1 — so the wait before the
// ds_store covers a load that has been in flight for a whole K-step.

#define OUT_F16       0
#define OUT_F16_RELU  1
#define OUT_F32       2
#define OUT_Q_BHSD    3   // f16, [((b*NH+h)*SEQ+s)*DK+d]
#define OUT_V_T       4   // f16, [((b*NH+h)*DK+d)*SEQ+s]
#define OUT_R_F32     5   // f32, [((b*NH+h)*SEQ+s)*DK+d]

__global__ void gemm_nt_kernel(const _Float16* __restrict__ A,
                               const _Float16* __restrict__ Wt,
                               const float* __restrict__ bias,
                               void* __restrict__ out,
                               int N, int K, int mode) {
  __shared__ __align__(64) _Float16 bsh[2][64 * 32];

  int tid = threadIdx.x;
  int wave = tid >> 5;
  int lane = tid & 31;
  int m0 = blockIdx.x * 128 + wave * 16;
  int n0 = blockIdx.y * 64;

  const _Float16* Arow = A + (size_t)m0 * K;
  // B staging: thread -> (row, 8-half chunk) of the 64x32 tile
  int srow = tid >> 2;
  int sch = (tid & 3) * 8;
  const _Float16* Wrow = Wt + (size_t)(n0 + srow) * K + sch;
  _Float16* sslot0 = &bsh[0][srow * 32 + sch];
  _Float16* sslot1 = &bsh[1][srow * 32 + sch];

  v8f acc[4] = {};

  // prologue: stage k=0 tile directly; start k=32 tile's global load;
  // load first A tile.
  *(v8h*)sslot0 = *(const v8h*)(Wrow);
  v8h g = {};
  if (K > 32) g = *(const v8h*)(Wrow + 32);
  v16h a = load_a_tile(Arow, K);
  __syncthreads();

  int cur = 0;
#pragma unroll 2
  for (int k0 = 0; k0 < K; k0 += 32) {
    v16h an = a;
    if (k0 + 32 < K) {
      // stage the in-flight tile (k0+32) into the other buffer
      *(v8h*)(cur ? sslot0 : sslot1) = g;
      // start global load of tile k0+64
      if (k0 + 64 < K) {
        __builtin_prefetch(Wrow + k0 + 96, 0, 3);
        g = *(const v8h*)(Wrow + k0 + 64);
      }
      an = load_a_tile(Arow + k0 + 32, K);
    }

    const _Float16* bb = &bsh[cur][0];
    v16h b0 = load_b_tile(bb + 0 * 16 * 32, 32);
    v16h b1 = load_b_tile(bb + 1 * 16 * 32, 32);
    v16h b2 = load_b_tile(bb + 2 * 16 * 32, 32);
    v16h b3 = load_b_tile(bb + 3 * 16 * 32, 32);
    acc[0] = wmma32(a, b0, acc[0]);
    acc[1] = wmma32(a, b1, acc[1]);
    acc[2] = wmma32(a, b2, acc[2]);
    acc[3] = wmma32(a, b3, acc[3]);

    a = an;
    __syncthreads();   // next buffer staged; current buffer free for reuse
    cur ^= 1;
  }

  int col = lane & 15;
  int rowoff = ((lane >> 4) & 1) * 8;
#pragma unroll
  for (int r = 0; r < 8; ++r) {
    int m = m0 + r + rowoff;
#pragma unroll
    for (int t = 0; t < 4; ++t) {
      int n = n0 + t * 16 + col;
      float v = acc[t][r] + bias[n];
      switch (mode) {
        case OUT_F16:
          ((_Float16*)out)[(size_t)m * N + n] = (_Float16)v; break;
        case OUT_F16_RELU:
          ((_Float16*)out)[(size_t)m * N + n] = (_Float16)fmaxf(v, 0.0f); break;
        case OUT_F32:
          ((float*)out)[(size_t)m * N + n] = v; break;
        case OUT_Q_BHSD: {
          int b = m >> 11, s = m & (SEQ - 1), h = n >> 6, d = n & 63;
          ((_Float16*)out)[(((size_t)b * NH + h) * SEQ + s) * DK + d] = (_Float16)v;
        } break;
        case OUT_V_T: {
          int b = m >> 11, s = m & (SEQ - 1), h = n >> 6, d = n & 63;
          ((_Float16*)out)[(((size_t)b * NH + h) * DK + d) * SEQ + s] = (_Float16)v;
        } break;
        case OUT_R_F32: {
          int b = m >> 11, s = m & (SEQ - 1), h = n >> 6, d = n & 63;
          ((float*)out)[(((size_t)b * NH + h) * SEQ + s) * DK + d] = v;
        } break;
      }
    }
  }
}

// ---------------- sequential tanh scan ----------------
// h_t = tanh(h_{t-1} * wdiag[h,d] + R[bh,t,d]),   wdiag = sum_k W_h[h,d,k]

__global__ void recurrence_kernel(const float* __restrict__ R,
                                  const float* __restrict__ W_h,
                                  _Float16* __restrict__ Hh) {
  int idx = blockIdx.x * blockDim.x + threadIdx.x;   // [0, 64*64)
  int bh = idx >> 6, d = idx & 63, h = bh & 7;
  float w = 0.0f;
  const float* wr = W_h + ((size_t)h * DK + d) * DK;
#pragma unroll 8
  for (int k = 0; k < DK; ++k) w += wr[k];
  float hv = 0.0f;
  const float* r = R + (size_t)bh * SEQ * DK + d;
  _Float16* o = Hh + (size_t)bh * SEQ * DK + d;
  for (int s = 0; s < SEQ; ++s) {
    hv = tanhf(hv * w + r[s * DK]);
    o[s * DK] = (_Float16)hv;
  }
}

// ---------------- flash attention ----------------
// grid (B*NH, SEQ/64), 128 threads = 4 waves; wave = one 16-row q-tile.
// scores = Q H^T /8 + (k>q ? 1 : 0); softmax over ALL keys; O = P V.
// H-tile loads for key-chunk kc+32 are issued before chunk kc's softmax.

__global__ void attention_kernel(const _Float16* __restrict__ Qh,
                                 const _Float16* __restrict__ Hh,
                                 const _Float16* __restrict__ Vt,
                                 _Float16* __restrict__ AttnH) {
  __shared__ __align__(64) _Float16 pbuf[4][16 * 32];

  int wave = threadIdx.x >> 5;
  int lane = threadIdx.x & 31;
  int bh = blockIdx.x;
  int b = bh >> 3, head = bh & 7;
  int q0 = blockIdx.y * 64 + wave * 16;

  const _Float16* Qbase = Qh + (size_t)bh * SEQ * DK;
  const _Float16* Hbase = Hh + (size_t)bh * SEQ * DK;
  const _Float16* Vbase = Vt + (size_t)bh * DK * SEQ;

  v16h qa0 = load_a_tile(Qbase + (size_t)q0 * DK +  0, DK);
  v16h qa1 = load_a_tile(Qbase + (size_t)q0 * DK + 32, DK);

  v8f o[4] = {};
  float rmax[8], rsum[8];
#pragma unroll
  for (int r = 0; r < 8; ++r) { rmax[r] = -1e30f; rsum[r] = 0.0f; }

  int half = (lane >> 4) & 1;
  int col = lane & 15;

  // prologue: H tiles of chunk 0
  v16h hb0 = load_b_tile(Hbase +  0, DK);
  v16h hb1 = load_b_tile(Hbase + 32, DK);
  v16h hb2 = load_b_tile(Hbase + (size_t)16 * DK +  0, DK);
  v16h hb3 = load_b_tile(Hbase + (size_t)16 * DK + 32, DK);

#pragma unroll 1
  for (int kc = 0; kc < SEQ; kc += 32) {
    // ---- scores for 32 keys (two 16-key N tiles, K=64 as 2x32) ----
    v8f sc0 = {}, sc1 = {};
    sc0 = wmma32(qa0, hb0, sc0);
    sc0 = wmma32(qa1, hb1, sc0);
    sc1 = wmma32(qa0, hb2, sc1);
    sc1 = wmma32(qa1, hb3, sc1);

    // pipeline: fetch next chunk's H tiles while softmax runs
    int kn = kc + 32;
    if (kn < SEQ) {
      hb0 = load_b_tile(Hbase + (size_t)kn * DK +  0, DK);
      hb1 = load_b_tile(Hbase + (size_t)kn * DK + 32, DK);
      hb2 = load_b_tile(Hbase + (size_t)(kn + 16) * DK +  0, DK);
      hb3 = load_b_tile(Hbase + (size_t)(kn + 16) * DK + 32, DK);
    }

#pragma unroll
    for (int r = 0; r < 8; ++r) {
      int qrow = q0 + r + 8 * half;
      int k0i = kc + col, k1i = kc + 16 + col;
      float s0 = sc0[r] * 0.125f + ((k0i > qrow) ? 1.0f : 0.0f);
      float s1 = sc1[r] * 0.125f + ((k1i > qrow) ? 1.0f : 0.0f);

      float cm = fmaxf(s0, s1);
#pragma unroll
      for (int msk = 1; msk < 16; msk <<= 1) cm = fmaxf(cm, __shfl_xor(cm, msk, 32));
      float nm = fmaxf(rmax[r], cm);
      float scale = __expf(rmax[r] - nm);
      rmax[r] = nm;

      float p0 = __expf(s0 - nm);
      float p1 = __expf(s1 - nm);
      float psum = p0 + p1;
#pragma unroll
      for (int msk = 1; msk < 16; msk <<= 1) psum += __shfl_xor(psum, msk, 32);
      rsum[r] = rsum[r] * scale + psum;

#pragma unroll
      for (int t = 0; t < 4; ++t) o[t][r] *= scale;

      pbuf[wave][(r + 8 * half) * 32 + col]      = (_Float16)p0;
      pbuf[wave][(r + 8 * half) * 32 + 16 + col] = (_Float16)p1;
    }
    __syncthreads();
    v16h pa = load_a_tile((const _Float16*)&pbuf[wave][0], 32);
    __syncthreads();

    // ---- O += P(16x32) * V(32keys x 64d) ----
#pragma unroll
    for (int t = 0; t < 4; ++t) {
      v16h bv = load_b_tile(Vbase + (size_t)(t * 16) * SEQ + kc, SEQ);
      o[t] = wmma32(pa, bv, o[t]);
    }
  }

  // ---- finalize: normalize and write (b, s, h*64+d) f16 ----
#pragma unroll
  for (int r = 0; r < 8; ++r) {
    float inv = 1.0f / rsum[r];
    int s = q0 + r + 8 * half;
#pragma unroll
    for (int t = 0; t < 4; ++t) {
      size_t idx = ((size_t)b * SEQ + s) * DM + head * DK + t * 16 + col;
      AttnH[idx] = (_Float16)(o[t][r] * inv);
    }
  }
}

// ---------------- layernorms (one wave per row) ----------------

__global__ void ln1_kernel(const float* __restrict__ x, const float* __restrict__ y,
                           const float* __restrict__ g, const float* __restrict__ be,
                           float* __restrict__ x1f, _Float16* __restrict__ x1h) {
  int wave = threadIdx.x >> 5, lane = threadIdx.x & 31;
  int m = blockIdx.x * 8 + wave;
  int b = m >> 11, s = m & (SEQ - 1);
  const float* xr = x + ((size_t)s * NB + b) * DM;
  const float* yr = y + (size_t)m * DM;
  float v[16], sum = 0.f, sq = 0.f;
#pragma unroll
  for (int i = 0; i < 16; ++i) {
    int d = i * 32 + lane;
    float t = xr[d] + yr[d];
    v[i] = t; sum += t; sq += t * t;
  }
#pragma unroll
  for (int msk = 16; msk >= 1; msk >>= 1) { sum += __shfl_xor(sum, msk, 32); sq += __shfl_xor(sq, msk, 32); }
  float mean = sum * (1.0f / DM);
  float var = sq * (1.0f / DM) - mean * mean;
  float rstd = rsqrtf(var + 1e-5f);
#pragma unroll
  for (int i = 0; i < 16; ++i) {
    int d = i * 32 + lane;
    float o = (v[i] - mean) * rstd * g[d] + be[d];
    x1f[(size_t)m * DM + d] = o;
    x1h[(size_t)m * DM + d] = (_Float16)o;
  }
}

__global__ void ln2_kernel(const float* __restrict__ x1f, const float* __restrict__ y2,
                           const float* __restrict__ g, const float* __restrict__ be,
                           float* __restrict__ out) {
  int wave = threadIdx.x >> 5, lane = threadIdx.x & 31;
  int m = blockIdx.x * 8 + wave;
  int b = m >> 11, s = m & (SEQ - 1);
  const float* xr = x1f + (size_t)m * DM;
  const float* yr = y2 + (size_t)m * DM;
  float v[16], sum = 0.f, sq = 0.f;
#pragma unroll
  for (int i = 0; i < 16; ++i) {
    int d = i * 32 + lane;
    float t = xr[d] + yr[d];
    v[i] = t; sum += t; sq += t * t;
  }
#pragma unroll
  for (int msk = 16; msk >= 1; msk >>= 1) { sum += __shfl_xor(sum, msk, 32); sq += __shfl_xor(sq, msk, 32); }
  float mean = sum * (1.0f / DM);
  float var = sq * (1.0f / DM) - mean * mean;
  float rstd = rsqrtf(var + 1e-5f);
#pragma unroll
  for (int i = 0; i < 16; ++i) {
    int d = i * 32 + lane;
    out[((size_t)s * NB + b) * DM + d] = (v[i] - mean) * rstd * g[d] + be[d];
  }
}

// ---------------- host launch ----------------

extern "C" void kernel_launch(void* const* d_in, const int* in_sizes, int n_in,
                              void* d_out, int out_size, void* d_ws, size_t ws_size,
                              hipStream_t stream) {
  const float* x     = (const float*)d_in[0];
  const float* Wq    = (const float*)d_in[1];
  const float* bq    = (const float*)d_in[2];
  const float* Wv    = (const float*)d_in[3];
  const float* bv    = (const float*)d_in[4];
  const float* Wr    = (const float*)d_in[5];
  const float* br    = (const float*)d_in[6];
  const float* W_h   = (const float*)d_in[7];
  const float* Wo    = (const float*)d_in[8];
  const float* bo    = (const float*)d_in[9];
  const float* ln1_g = (const float*)d_in[10];
  const float* ln1_b = (const float*)d_in[11];
  const float* Wf1   = (const float*)d_in[12];
  const float* bf1   = (const float*)d_in[13];
  const float* Wf2   = (const float*)d_in[14];
  const float* bf2   = (const float*)d_in[15];
  const float* ln2_g = (const float*)d_in[16];
  const float* ln2_b = (const float*)d_in[17];
  float* out = (float*)d_out;
  char* ws = (char*)d_ws;

  size_t off = 0;
  auto alloc = [&](size_t bytes) { char* p = ws + off; off += (bytes + 255) & ~(size_t)255; return p; };

  _Float16* XH   = (_Float16*)alloc((size_t)MROWS * DM * 2);   // later reused as AttnH
  _Float16* WQH  = (_Float16*)alloc((size_t)DM * DM * 2);
  _Float16* WVH  = (_Float16*)alloc((size_t)DM * DM * 2);
  _Float16* WRH  = (_Float16*)alloc((size_t)DM * DM * 2);
  _Float16* WOH  = (_Float16*)alloc((size_t)DM * DM * 2);
  _Float16* WF1T = (_Float16*)alloc((size_t)DFF * DM * 2);
  _Float16* WF2T = (_Float16*)alloc((size_t)DM * DFF * 2);
  _Float16* QH   = (_Float16*)alloc((size_t)MROWS * DM * 2);   // later reused as X1H
  _Float16* VT   = (_Float16*)alloc((size_t)MROWS * DM * 2);
  float*    RF   = (float*)   alloc((size_t)MROWS * DM * 4);   // later reused as Y1, Y2
  _Float16* HH   = (_Float16*)alloc((size_t)MROWS * DM * 2);
  float*    X1F  = (float*)   alloc((size_t)MROWS * DM * 4);
  _Float16* FFH  = (_Float16*)alloc((size_t)MROWS * DFF * 2);
  _Float16* ATTNH = XH;
  _Float16* X1H   = QH;
  float*    Y1    = RF;
  float*    Y2    = RF;

  // 1) pack / convert
  pack_x_kernel<<<(MROWS * DM) / 256, 256, 0, stream>>>(x, XH);
  cvt_kernel<<<(DM * DM) / 256, 256, 0, stream>>>(Wq, WQH, DM * DM);
  cvt_kernel<<<(DM * DM) / 256, 256, 0, stream>>>(Wv, WVH, DM * DM);
  cvt_kernel<<<(DM * DM) / 256, 256, 0, stream>>>(Wr, WRH, DM * DM);
  cvt_kernel<<<(DM * DM) / 256, 256, 0, stream>>>(Wo, WOH, DM * DM);
  cvtT_kernel<<<(DM * DFF) / 256, 256, 0, stream>>>(Wf1, WF1T, DM, DFF);
  cvtT_kernel<<<(DFF * DM) / 256, 256, 0, stream>>>(Wf2, WF2T, DFF, DM);

  // 2) projections (WMMA)
  dim3 gP(MROWS / 128, DM / 64);
  gemm_nt_kernel<<<gP, 256, 0, stream>>>(XH, WQH, bq, QH, DM, DM, OUT_Q_BHSD);
  gemm_nt_kernel<<<gP, 256, 0, stream>>>(XH, WVH, bv, VT, DM, DM, OUT_V_T);
  gemm_nt_kernel<<<gP, 256, 0, stream>>>(XH, WRH, br, RF, DM, DM, OUT_R_F32);

  // 3) sequential tanh scan -> H (f16, (bh, s, d))
  recurrence_kernel<<<16, 256, 0, stream>>>(RF, W_h, HH);

  // 4) flash attention (WMMA) -> AttnH (reuses XH)
  attention_kernel<<<dim3(NB * NH, SEQ / 64), 128, 0, stream>>>(QH, HH, VT, ATTNH);

  // 5) Wo projection -> Y1 f32 (reuses RF)
  gemm_nt_kernel<<<gP, 256, 0, stream>>>(ATTNH, WOH, bo, Y1, DM, DM, OUT_F32);

  // 6) residual + LN1 -> X1F f32, X1H f16 (reuses QH)
  ln1_kernel<<<MROWS / 8, 256, 0, stream>>>(x, Y1, ln1_g, ln1_b, X1F, X1H);

  // 7) FFN (WMMA)
  dim3 gF1(MROWS / 128, DFF / 64);
  gemm_nt_kernel<<<gF1, 256, 0, stream>>>(X1H, WF1T, bf1, FFH, DFF, DM, OUT_F16_RELU);
  gemm_nt_kernel<<<gP, 256, 0, stream>>>(FFH, WF2T, bf2, Y2, DM, DFF, OUT_F32);

  // 8) residual + LN2 -> out (S,B,D) f32
  ln2_kernel<<<MROWS / 8, 256, 0, stream>>>(X1F, Y2, ln2_g, ln2_b, out);
}
